// MTHARS_61091614819053
// MI455X (gfx1250) — matrix operand order repaired and checked by reading, not compiled, not run on hardware
//
#include <hip/hip_runtime.h>
#include <hip/hip_bf16.h>
#include <math.h>

// ---------- CDNA5 WMMA vector types ----------
typedef _Float16 v16h __attribute__((ext_vector_type(16)));
typedef _Float16 v8h  __attribute__((ext_vector_type(8)));
typedef float    v8f  __attribute__((ext_vector_type(8)));

// ---------- problem constants (from reference shapes) ----------
#define BATCH 16
#define NIN   6144
#define NSEQ  2048          // after stride-3 conv
#define C1    64
#define NCOL  (BATCH*NSEQ)  // 32768 GEMM columns
#define KDIM  1536          // 6*256 pooled channels
#define CLS_TOTAL (BATCH*12*NSEQ)

// =====================================================================
// layer1: conv1d(6->64, k=5, stride=3, pad=1) + BN + ReLU
// =====================================================================
__global__ void layer1_kernel(const float* __restrict__ x, const float* __restrict__ w,
                              const float* __restrict__ bias,
                              const float* __restrict__ g, const float* __restrict__ bb,
                              const float* __restrict__ m, const float* __restrict__ v,
                              float* __restrict__ out) {
  int idx = blockIdx.x * blockDim.x + threadIdx.x;       // B*64*2048 = 2^21
  if (idx >= BATCH * C1 * NSEQ) return;
  int n = idx & (NSEQ - 1);
  int o = (idx >> 11) & 63;
  int b = idx >> 17;
  float acc = bias[o];
  int p0 = n * 3 - 1;
#pragma unroll
  for (int c = 0; c < 6; ++c) {
    const float* xr = x + (b * 6 + c) * NIN;
    const float* wr = w + (o * 6 + c) * 5;
#pragma unroll
    for (int k = 0; k < 5; ++k) {
      int p = p0 + k;
      if (p >= 0 && p < NIN) acc += xr[p] * wr[k];
    }
  }
  float y = (acc - m[o]) * g[o] * rsqrtf(v[o] + 1e-5f) + bb[o];
  out[idx] = fmaxf(y, 0.f);
}

// =====================================================================
// SK-conv pieces
// =====================================================================
struct SKBranchParams {
  const float* w[3];
  const float* bias[3];
  const float* g[3];
  const float* bb[3];
  const float* m[3];
  const float* v[3];
};

struct SKAttnParams {
  const float* wfc;
  const float* fg; const float* fb; const float* fm; const float* fv;
  const float* wa[3];
  const float* ba[3];
};

// one block per (b,o); 256 threads sweep N; also block-reduce GAP
template <int O, int CING>
__global__ void sk_branch_kernel(const float* __restrict__ in, SKBranchParams p,
                                 float* __restrict__ feats,   // [3][B][O][N]
                                 float* __restrict__ s) {     // [B][O]
  const int CIN = 32 * CING;
  __shared__ float red[256];
  int bo = blockIdx.x;
  int b = bo / O, o = bo % O;
  int cbase = (o / (O / 32)) * CING;
  int t = threadIdx.x;
  float sl = 0.f;
  for (int rep = 0; rep < NSEQ / 256; ++rep) {
    int n = rep * 256 + t;
#pragma unroll
    for (int i = 0; i < 3; ++i) {
      const int d = i + 1;
      float acc = p.bias[i][o];
#pragma unroll
      for (int c = 0; c < CING; ++c) {
        const float* xr = in + ((size_t)(b * CIN + cbase + c)) * NSEQ;
        const float* wr = p.w[i] + (o * CING + c) * 3;
#pragma unroll
        for (int k = 0; k < 3; ++k) {
          int pos = n + (k - 1) * d;
          if (pos >= 0 && pos < NSEQ) acc += xr[pos] * wr[k];
        }
      }
      float y = (acc - p.m[i][o]) * p.g[i][o] * rsqrtf(p.v[i][o] + 1e-5f) + p.bb[i][o];
      y = fmaxf(y, 0.f);
      feats[(((size_t)i * BATCH + b) * O + o) * NSEQ + n] = y;
      sl += y;
    }
  }
  red[t] = sl; __syncthreads();
  for (int off = 128; off > 0; off >>= 1) {
    if (t < off) red[t] += red[t + off];
    __syncthreads();
  }
  if (t == 0) s[bo] = red[0] * (1.f / (float)NSEQ);
}

// one block per batch; blockDim = O
template <int O, int D>
__global__ void sk_attn_kernel(const float* __restrict__ s, SKAttnParams p,
                               float* __restrict__ attn) {  // [3][B][O]
  __shared__ float ssh[O];
  __shared__ float zsh[D];
  int b = blockIdx.x, t = threadIdx.x;
  ssh[t] = s[b * O + t];
  __syncthreads();
  if (t < D) {
    float acc = 0.f;
    for (int c = 0; c < O; ++c) acc += p.wfc[t * O + c] * ssh[c];
    float z = (acc - p.fm[t]) * p.fg[t] * rsqrtf(p.fv[t] + 1e-5f) + p.fb[t];
    zsh[t] = fmaxf(z, 0.f);
  }
  __syncthreads();
  float wv[3];
#pragma unroll
  for (int i = 0; i < 3; ++i) {
    float acc = p.ba[i][t];
#pragma unroll
    for (int j = 0; j < D; ++j) acc += p.wa[i][t * D + j] * zsh[j];
    wv[i] = acc;
  }
  float mx = fmaxf(wv[0], fmaxf(wv[1], wv[2]));
  float e0 = __expf(wv[0] - mx), e1 = __expf(wv[1] - mx), e2 = __expf(wv[2] - mx);
  float inv = 1.f / (e0 + e1 + e2);
  attn[(0 * BATCH + b) * O + t] = e0 * inv;
  attn[(1 * BATCH + b) * O + t] = e1 * inv;
  attn[(2 * BATCH + b) * O + t] = e2 * inv;
}

template <int O>
__global__ void sk_apply_kernel(const float* __restrict__ feats, const float* __restrict__ attn,
                                float* __restrict__ out) {
  const size_t P = (size_t)BATCH * O * NSEQ;
  size_t idx = (size_t)blockIdx.x * blockDim.x + threadIdx.x;
  if (idx >= P) return;
  int bo = (int)(idx >> 11);
  int o = bo % O, b = bo / O;
  float a0 = attn[(0 * BATCH + b) * O + o];
  float a1 = attn[(1 * BATCH + b) * O + o];
  float a2 = attn[(2 * BATCH + b) * O + o];
  out[idx] = feats[idx] * a0 + feats[P + idx] * a1 + feats[2 * P + idx] * a2;
}

// =====================================================================
// per-row inclusive prefix sum (cs[row][0]=0, cs[row][1..N]=cumsum)
// =====================================================================
__global__ void scan_kernel(const float* __restrict__ in, float* __restrict__ cs) {
  __shared__ float sd[256];
  int row = blockIdx.x;                 // B*256 rows
  int t = threadIdx.x;
  const float* xr = in + (size_t)row * NSEQ;
  float* cr = cs + (size_t)row * (NSEQ + 1);
  if (t == 0) cr[0] = 0.f;
  float running = 0.f;
  for (int ch = 0; ch < NSEQ / 256; ++ch) {
    sd[t] = xr[ch * 256 + t];
    __syncthreads();
    for (int off = 1; off < 256; off <<= 1) {
      float add = (t >= off) ? sd[t - off] : 0.f;
      __syncthreads();
      sd[t] += add;
      __syncthreads();
    }
    cr[1 + ch * 256 + t] = running + sd[t];
    running += sd[255];
    __syncthreads();
  }
}

// =====================================================================
// build f16 pooled feature, column-major-for-GEMM: feat16[j][k], j=b*N+n
// =====================================================================
__global__ void feat_build_kernel(const float* __restrict__ cs, _Float16* __restrict__ feat16,
                                  int L0, int L1, int L2, int L3, int L4, int L5) {
  size_t idx = (size_t)blockIdx.x * blockDim.x + threadIdx.x;
  if (idx >= (size_t)NCOL * KDIM) return;
  int k = (int)(idx % KDIM);
  int j = (int)(idx / KDIM);
  int si = k >> 8, c = k & 255;
  int b = j >> 11, n = j & (NSEQ - 1);
  int L = si == 0 ? L0 : si == 1 ? L1 : si == 2 ? L2 : si == 3 ? L3 : si == 4 ? L4 : L5;
  int half = L >> 1;
  int st = min(max(n - half, 0), NSEQ);
  int en = min(max(n - half + L, 0), NSEQ);
  const float* cr = cs + (size_t)(b * 256 + c) * (NSEQ + 1);
  feat16[idx] = (_Float16)((cr[en] - cr[st]) / (float)L);
}

// pack head weights: rows 0..11 = wc, 12..13 = ws, 14..15 = 0 ; f16 [16][1536]
__global__ void pack_w_kernel(const float* __restrict__ wc, const float* __restrict__ ws,
                              _Float16* __restrict__ W16) {
  int idx = blockIdx.x * blockDim.x + threadIdx.x;     // 16*1536
  if (idx >= 16 * KDIM) return;
  int k = idx % KDIM, mm = idx / KDIM;
  float vv = 0.f;
  if (mm < 12) vv = wc[mm * KDIM + k];
  else if (mm < 14) vv = ws[(mm - 12) * KDIM + k];
  W16[idx] = (_Float16)vv;
}

// =====================================================================
// head GEMM: out[16,32768] = W16[16,1536] x feat16[1536,32768]
// one wave per 16-column tile, 48x v_wmma_f32_16x16x32_f16, f32 acc
// =====================================================================
__global__ void head_gemm_kernel(const _Float16* __restrict__ W16,
                                 const _Float16* __restrict__ feat16,
                                 const float* __restrict__ bc, const float* __restrict__ bs,
                                 float* __restrict__ out) {
  int wave = (blockIdx.x * blockDim.x + threadIdx.x) >> 5;  // 2048 waves
  int lane = threadIdx.x & 31;
  int jbase = wave * 16;
  int lm = lane & 15;
  int hi = lane >> 4;
  v8f c = {};
  const _Float16* arow = W16 + lm * KDIM + hi * 16;                   // A: 16-bit 16x32 layout
  const _Float16* brow = feat16 + (size_t)(jbase + lm) * KDIM + hi * 16;  // B: 16-bit 32x16 layout
#pragma unroll 4
  for (int kt = 0; kt < KDIM / 32; ++kt) {
    if (kt + 1 < KDIM / 32)
      __builtin_prefetch(brow + (kt + 1) * 32, 0, 3);                 // global_prefetch_b8
    v8h alo = *(const v8h*)(arow + kt * 32);
    v8h ahi = *(const v8h*)(arow + kt * 32 + 8);
    v16h a;
#pragma unroll
    for (int i = 0; i < 8; ++i) { a[i] = alo[i]; a[8 + i] = ahi[i]; }
    v16h bfrag = *(const v16h*)(brow + kt * 32);
    c = __builtin_amdgcn_wmma_f32_16x16x32_f16(false, a, false, bfrag,
                                               (short)0, c, false, false);
  }
  int j = jbase + lm;
  int b = j >> 11, n = j & (NSEQ - 1);
#pragma unroll
  for (int r = 0; r < 8; ++r) {
    int mrow = hi * 8 + r;
    float val = c[r];
    if (mrow < 12)       out[((size_t)b * 12 + mrow) * NSEQ + n] = val + bc[mrow];
    else if (mrow < 14)  out[CLS_TOTAL + ((size_t)b * 2 + (mrow - 12)) * NSEQ + n] = val + bs[mrow - 12];
  }
}

// =====================================================================
// host launch
// =====================================================================
extern "C" void kernel_launch(void* const* d_in, const int* in_sizes, int n_in,
                              void* d_out, int out_size, void* d_ws, size_t ws_size,
                              hipStream_t stream) {
  int i = 0;
  const float* x = (const float*)d_in[i++];            // (16,6,6144)
  // layer1
  const float* l1w = (const float*)d_in[i++];
  const float* l1b = (const float*)d_in[i++];
  const float* l1g = (const float*)d_in[i++];
  const float* l1bb = (const float*)d_in[i++];
  const float* l1m = (const float*)d_in[i++];
  const float* l1v = (const float*)d_in[i++];
  // sk blocks share layout: w[3], b[3], bn[3]{g,b,m,v}, wfc, bnfc{g,b,m,v}, wa[3], ba[3]
  SKBranchParams bp1{}, bp2{};
  SKAttnParams   ap1{}, ap2{};
  for (int s = 0; s < 2; ++s) {
    SKBranchParams& bp = s ? bp2 : bp1;
    SKAttnParams&   ap = s ? ap2 : ap1;
    for (int j = 0; j < 3; ++j) bp.w[j] = (const float*)d_in[i++];
    for (int j = 0; j < 3; ++j) bp.bias[j] = (const float*)d_in[i++];
    for (int j = 0; j < 3; ++j) {
      bp.g[j] = (const float*)d_in[i++];
      bp.bb[j] = (const float*)d_in[i++];
      bp.m[j] = (const float*)d_in[i++];
      bp.v[j] = (const float*)d_in[i++];
    }
    ap.wfc = (const float*)d_in[i++];
    ap.fg = (const float*)d_in[i++];
    ap.fb = (const float*)d_in[i++];
    ap.fm = (const float*)d_in[i++];
    ap.fv = (const float*)d_in[i++];
    for (int j = 0; j < 3; ++j) ap.wa[j] = (const float*)d_in[i++];
    for (int j = 0; j < 3; ++j) ap.ba[j] = (const float*)d_in[i++];
  }
  const float* wc = (const float*)d_in[i++];   // (12,1536,1)
  const float* bc = (const float*)d_in[i++];
  const float* wsw = (const float*)d_in[i++];  // (2,1536,1)
  const float* bs = (const float*)d_in[i++];

  // ---- workspace layout (floats), feat16 overlays dead branch-feature region ----
  float* h1 = (float*)d_ws;                       // 16*64*2048
  float* h2 = h1 + (size_t)BATCH * 64 * NSEQ;     // 16*128*2048
  float* h3 = h2 + (size_t)BATCH * 128 * NSEQ;    // 16*256*2048
  float* cs = h3 + (size_t)BATCH * 256 * NSEQ;    // 16*256*2049
  float* feats = cs + (size_t)BATCH * 256 * (NSEQ + 1);  // up to 3*16*256*2048
  float* sbuf = feats + (size_t)3 * BATCH * 256 * NSEQ;  // 4096
  float* attn = sbuf + 4096;                      // 12288
  _Float16* W16 = (_Float16*)(attn + 12288);      // 16*1536 halves
  _Float16* feat16 = (_Float16*)feats;            // overlay (feats dead by then)

  // multi-scale window lengths (same floor/sqrt formula as reference, N=2048)
  const double sc[3] = {0.2, 0.5, 0.8};
  int L[6];
  for (int s = 0; s < 3; ++s) {
    L[2 * s]     = (int)floor((double)NSEQ * sqrt(sc[s]));
    L[2 * s + 1] = (int)floor((double)NSEQ / sqrt(sc[s]));
  }

  // ---- pipeline ----
  layer1_kernel<<<(BATCH * C1 * NSEQ) / 256, 256, 0, stream>>>(
      x, l1w, l1b, l1g, l1bb, l1m, l1v, h1);

  sk_branch_kernel<128, 2><<<BATCH * 128, 256, 0, stream>>>(h1, bp1, feats, sbuf);
  sk_attn_kernel<128, 4><<<BATCH, 128, 0, stream>>>(sbuf, ap1, attn);
  sk_apply_kernel<128><<<(BATCH * 128 * NSEQ) / 256, 256, 0, stream>>>(feats, attn, h2);

  sk_branch_kernel<256, 4><<<BATCH * 256, 256, 0, stream>>>(h2, bp2, feats, sbuf);
  sk_attn_kernel<256, 8><<<BATCH, 256, 0, stream>>>(sbuf, ap2, attn);
  sk_apply_kernel<256><<<(BATCH * 256 * NSEQ) / 256, 256, 0, stream>>>(feats, attn, h3);

  scan_kernel<<<BATCH * 256, 256, 0, stream>>>(h3, cs);
  pack_w_kernel<<<(16 * KDIM) / 256, 256, 0, stream>>>(wc, wsw, W16);
  feat_build_kernel<<<(int)(((size_t)NCOL * KDIM) / 256), 256, 0, stream>>>(
      cs, feat16, L[0], L[1], L[2], L[3], L[4], L[5]);

  head_gemm_kernel<<<NCOL / 16 / 8, 256, 0, stream>>>(W16, feat16, bc, bs, (float*)d_out);
}
